// SNNetwork_40132174414299
// MI455X (gfx1250) — matrix operand order repaired and checked by reading, not compiled, not run on hardware
//
#include <hip/hip_runtime.h>
#include <hip/hip_fp16.h>
#include <stdint.h>

typedef __attribute__((ext_vector_type(16))) _Float16 v16h;
typedef __attribute__((ext_vector_type(8)))  float    v8f;

#define SEQ   10
#define TPB   128           // 4 waves; LDS ~42KB/WG -> ~7 WGs per WGP
#define NWAVE (TPB / 32)

// ---- exact JAX threefry2x32 with key(42) -> (k1=0, k2=42) ----
__device__ __forceinline__ void tf_round(uint32_t& x0, uint32_t& x1, int r) {
  x0 += x1;
  x1 = (x1 << r) | (x1 >> (32 - r));   // -> v_alignbit_b32
  x1 ^= x0;
}

__device__ __forceinline__ uint2 threefry2x32_42(uint32_t c0, uint32_t c1) {
  const uint32_t ks0 = 0u, ks1 = 42u, ks2 = 0u ^ 42u ^ 0x1BD11BDAu;
  uint32_t x0 = c0 + ks0, x1 = c1 + ks1;
  tf_round(x0,x1,13); tf_round(x0,x1,15); tf_round(x0,x1,26); tf_round(x0,x1,6);
  x0 += ks1; x1 += ks2 + 1u;
  tf_round(x0,x1,17); tf_round(x0,x1,29); tf_round(x0,x1,16); tf_round(x0,x1,24);
  x0 += ks2; x1 += ks0 + 2u;
  tf_round(x0,x1,13); tf_round(x0,x1,15); tf_round(x0,x1,26); tf_round(x0,x1,6);
  x0 += ks0; x1 += ks1 + 3u;
  tf_round(x0,x1,17); tf_round(x0,x1,29); tf_round(x0,x1,16); tf_round(x0,x1,24);
  x0 += ks1; x1 += ks2 + 4u;
  tf_round(x0,x1,13); tf_round(x0,x1,15); tf_round(x0,x1,26); tf_round(x0,x1,6);
  x0 += ks2; x1 += ks0 + 5u;
  return make_uint2(x0, x1);
}

// jax.random.uniform bit pattern: bitcast((bits>>9)|0x3f800000) - 1.0
__device__ __forceinline__ float bits_to_unit(uint32_t b) {
  return __uint_as_float((b >> 9) | 0x3f800000u) - 1.0f;
}

__global__ __launch_bounds__(TPB) void snn_lif_wmma_kernel(
    const float* __restrict__ x,       // [B,12]
    const float* __restrict__ w_in0,   // [6,12]
    const float* __restrict__ w_rec0,  // [6,6]
    const float* __restrict__ w_in1,   // [1,6]
    const float* __restrict__ w_rec1,  // [1,1]
    float* __restrict__ out,           // [B]
    unsigned B)
{
  // per-wave staging: spikes [t][32 elems][16 ch (12 used)], z feedback [2 tiles][16 b][8 n]
  __shared__ __align__(16) __half s_spk[NWAVE][SEQ][32][16];
  __shared__ __align__(16) __half s_z[NWAVE][2][16][8];

  const int tid = threadIdx.x;
  const int w   = tid >> 5;      // wave in block
  const int q   = tid & 31;      // lane
  const int h   = q >> 4;        // K-half (A/B layouts) == tile index for this lane's element
  const int n   = q & 15;        // column N (B/C/D layouts) == row-in-tile for this lane's element
  const unsigned e = blockIdx.x * TPB + tid;   // batch element owned by this lane

  // ---- WMMA B fragment: Wcat[K=32][N=16] f16, K 0-11 = w_in0[n][k], 12-17 = w_rec0[n][k-12]
  // layout: elem j, lanes 0-15 hold K=j (j<8), 16+j-8 (j>=8) ; lanes 16-31 hold K=8+j (j<8), 24+j-8 (j>=8)
  union { v16h v; uint32_t u[8]; __half hh[16]; } bfrag;
  #pragma unroll
  for (int j = 0; j < 8; ++j) bfrag.u[j] = 0u;
  if (n < 6) {
    #pragma unroll
    for (int j = 0; j < 16; ++j) {
      int k = (j < 8) ? (8 * h + j) : (16 + 8 * h + (j - 8));
      float wv = 0.0f;
      if (k < 12)      wv = w_in0[n * 12 + k];
      else if (k < 18) wv = w_rec0[n * 6 + (k - 12)];
      bfrag.hh[j] = __float2half(wv);
    }
  }

  float win1[6];
  #pragma unroll
  for (int j = 0; j < 6; ++j) win1[j] = w_in1[j];
  const float wr1 = w_rec1[0];

  // zero pad channels 12..15 of this lane's spike rows (all 10 t)
  #pragma unroll
  for (int t = 0; t < SEQ; ++t)
    *reinterpret_cast<uint2*>(&s_spk[w][t][q][12]) = make_uint2(0u, 0u);

  // stream this element's 12 inputs (48B, coalesced, 16B aligned)
  const float4* xr4 = reinterpret_cast<const float4*>(x + (size_t)e * 12);
  float4 xa = xr4[0], xb = xr4[1], xc = xr4[2];
  float xv[12] = { xa.x,xa.y,xa.z,xa.w, xb.x,xb.y,xb.z,xb.w, xc.x,xc.y,xc.z,xc.w };

  // ---- Poisson encoding, bit-exact threefry: pair (t, t+5) per call
  const uint32_t tstr  = 12u * B;        // stride of t in flat index
  const uint32_t halfN = 5u * tstr;      // N/2
  #pragma unroll
  for (int t = 0; t < 5; ++t) {
    uint32_t r0[6], r5[6];               // packed f16 spike rows (1.0 = 0x3C00)
    #pragma unroll
    for (int c2 = 0; c2 < 6; ++c2) {
      const int c = 2 * c2;
      uint32_t f0 = (uint32_t)t * tstr + e * 12u + (uint32_t)c;
      uint2 ra = threefry2x32_42(f0,      f0      + halfN);
      uint2 rb = threefry2x32_42(f0 + 1u, f0 + 1u + halfN);
      r0[c2] = (bits_to_unit(ra.x) < xv[c]   ? 0x3C00u     : 0u)
             | (bits_to_unit(rb.x) < xv[c+1] ? 0x3C000000u : 0u);
      r5[c2] = (bits_to_unit(ra.y) < xv[c]   ? 0x3C00u     : 0u)
             | (bits_to_unit(rb.y) < xv[c+1] ? 0x3C000000u : 0u);
    }
    uint32_t* d0 = reinterpret_cast<uint32_t*>(&s_spk[w][t][q][0]);
    uint32_t* d5 = reinterpret_cast<uint32_t*>(&s_spk[w][t + 5][q][0]);
    *reinterpret_cast<uint4*>(d0)     = make_uint4(r0[0], r0[1], r0[2], r0[3]);
    *reinterpret_cast<uint2*>(d0 + 4) = make_uint2(r0[4], r0[5]);
    *reinterpret_cast<uint4*>(d5)     = make_uint4(r5[0], r5[1], r5[2], r5[3]);
    *reinterpret_cast<uint2*>(d5 + 4) = make_uint2(r5[4], r5[5]);
  }

  // zero z feedback buffer (2048B total, 16B per thread)
  reinterpret_cast<uint4*>(&s_z[0][0][0][0])[tid] = make_uint4(0u, 0u, 0u, 0u);
  __syncthreads();

  // layer-0 state in WMMA C/D layout: slot (lane q, vgpr r) = (batch M=r+8h, neuron N=n)
  v8f iacc[2], vacc[2];
  #pragma unroll
  for (int tl = 0; tl < 2; ++tl)
    #pragma unroll
    for (int r = 0; r < 8; ++r) { iacc[tl][r] = 0.0f; vacc[tl][r] = 0.0f; }

  float z1 = 0.0f, v1 = 0.0f, i1 = 0.0f;

  for (int t = 0; t < SEQ; ++t) {
    #pragma unroll
    for (int tl = 0; tl < 2; ++tl) {
      // elementwise decay / threshold / reset (expression forms match reference)
      v8f idec;
      float zf[8];
      #pragma unroll
      for (int r = 0; r < 8; ++r) {
        float iv = iacc[tl][r], vv = vacc[tl][r];
        float vd = vv + 0.1f * (iv - vv);         // v + dt*tau_mem_inv*(-v+i)
        idec[r]  = iv - 0.2f * iv;                // i - dt*tau_syn_inv*i
        float zz = (vd - 1.0f) > 0.0f ? 1.0f : 0.0f;
        vacc[tl][r] = (1.0f - zz) * vd;
        zf[r] = zz;
      }
      // A fragment 16x32 f16: K 0-11 = spikes(t), K 12-17 = z_old, rest 0
      // lanes 0-15: K 0-7 (spk) + 16,17 (z4,z5); lanes 16-31: K 8-11 (spk) + 12-15 (z0..z3)
      union { v16h v; uint32_t u[8]; } afrag;
      const uint32_t* srow = reinterpret_cast<const uint32_t*>(&s_spk[w][t][tl * 16 + n][0]);
      uint4 s4 = *reinterpret_cast<const uint4*>(srow + 4 * h);
      const uint32_t* zrow = reinterpret_cast<const uint32_t*>(&s_z[w][tl][n][0]);
      uint2 z2 = *reinterpret_cast<const uint2*>(zrow + (h ? 0 : 2));
      afrag.u[0] = s4.x;
      afrag.u[1] = s4.y;
      afrag.u[2] = h ? z2.x : s4.z;
      afrag.u[3] = h ? z2.y : s4.w;
      afrag.u[4] = h ? 0u : z2.x;      // K=16,17 -> (z4,z5) for lanes 0-15
      afrag.u[5] = h ? 0u : z2.y;      // K=18,19 -> zero pad
      afrag.u[6] = 0u;
      afrag.u[7] = 0u;
      // i_new = [x_t | z_old] * Wcat^T + i_dec   (one WMMA, f32 accumulate)
      iacc[tl] = __builtin_amdgcn_wmma_f32_16x16x32_f16(
          false, afrag.v, false, bfrag.v, (short)0, idec, false, false);
      // publish z_new (f16) for next step's recurrence + layer 1
      if (n < 8) {
        #pragma unroll
        for (int r = 0; r < 8; ++r)
          s_z[w][tl][r + 8 * h][n] = __float2half(zf[r]);
      }
    }
    __syncthreads();

    // ---- layer 1, per-lane scalar (this lane's element is tile h, row n)
    const __half* zr = &s_z[w][h][n][0];
    float dot = 0.0f;
    #pragma unroll
    for (int j = 0; j < 6; ++j) dot += __half2float(zr[j]) * win1[j];
    float v1d = v1 + 0.1f * (i1 - v1);
    float i1n = (i1 - 0.2f * i1) + dot + z1 * wr1;   // uses OLD z1
    float z1n = (v1d - 1.0f) > 0.0f ? 1.0f : 0.0f;
    v1 = (1.0f - z1n) * v1d;
    z1 = z1n;
    i1 = i1n;
  }

  out[e] = z1;
}

extern "C" void kernel_launch(void* const* d_in, const int* in_sizes, int n_in,
                              void* d_out, int out_size, void* d_ws, size_t ws_size,
                              hipStream_t stream) {
  (void)n_in; (void)out_size; (void)d_ws; (void)ws_size;
  const float* x      = (const float*)d_in[0];
  const float* w_in0  = (const float*)d_in[1];
  const float* w_rec0 = (const float*)d_in[2];
  const float* w_in1  = (const float*)d_in[3];
  const float* w_rec1 = (const float*)d_in[4];
  float* out = (float*)d_out;
  unsigned B = (unsigned)(in_sizes[0] / 12);   // 1048576; divisible by TPB
  unsigned grid = B / TPB;
  snn_lif_wmma_kernel<<<grid, TPB, 0, stream>>>(x, w_in0, w_rec0, w_in1, w_rec1, out, B);
}
